// simple_attn_81106162418425
// MI455X (gfx1250) — compile-verified, matrix operand order checked by reading, not run on hardware
//
#include <hip/hip_runtime.h>
#include <cmath>

typedef __attribute__((ext_vector_type(16))) __bf16 v16bf;
typedef __attribute__((ext_vector_type(8)))  __bf16 v8bf;
typedef __attribute__((ext_vector_type(8)))  float  v8f;

#define B_      4
#define NSEQ    65536
#define CIN     192
#define HEADS   6
#define DH      32
#define TOKENS  (B_ * NSEQ)         // 262144
#define CQKV    (3 * CIN)           // 576
#define LN_EPS  1e-5f

#define WMMA_BF16(a, b, c) \
  __builtin_amdgcn_wmma_f32_16x16x32_bf16(false, (a), false, (b), (short)0, (c), false, false)

// Async global->LDS DMA (ASYNCcnt-tracked), 16 bytes per lane.
// vdst = 32-bit LDS byte address, vaddr = 64-bit global address.
#define ASYNC_G2L_B128(ldsaddr, gptr)                                   \
  asm volatile("global_load_async_to_lds_b128 %0, %1, off"              \
               :: "v"(ldsaddr), "v"(gptr) : "memory")
#define WAIT_ASYNC0() asm volatile("s_wait_asynccnt 0x0" ::: "memory")

__device__ __forceinline__ unsigned lds_addr_of(const void* p) {
  // Low 32 bits of the generic address of a __shared__ object are the LDS
  // byte offset (ISA 10.2 aperture mapping).
  return (unsigned)(size_t)p;
}

__device__ __forceinline__ v16bf pack16(v8bf lo, v8bf hi) {
  v16bf r;
#pragma unroll
  for (int i = 0; i < 8; ++i) { r[i] = lo[i]; r[i + 8] = hi[i]; }
  return r;
}

__device__ __forceinline__ v8bf cvt8(float4 a, float4 b) {
  v8bf r;
  r[0] = (__bf16)a.x; r[1] = (__bf16)a.y; r[2] = (__bf16)a.z; r[3] = (__bf16)a.w;
  r[4] = (__bf16)b.x; r[5] = (__bf16)b.y; r[6] = (__bf16)b.z; r[7] = (__bf16)b.w;
  return r;
}

// ---------------------------------------------------------------------------
// Utility kernels
// ---------------------------------------------------------------------------
__global__ void f32_to_bf16_kernel(const float* __restrict__ s, __bf16* __restrict__ d, int n) {
  int i = blockIdx.x * blockDim.x + threadIdx.x;
  if (i < n) d[i] = (__bf16)s[i];
}

__global__ void zero_f32_kernel(float* __restrict__ p, int n) {
  int i = blockIdx.x * blockDim.x + threadIdx.x;
  if (i < n) p[i] = 0.0f;
}

// ---------------------------------------------------------------------------
// Kernel 1: QKV projection (x f32 -> WMMA bf16) + bias + LayerNorm(k,v) epilogue
// Tile: 128 tokens x 64 out-cols, 8 waves of (2x2) 16x16 WMMA tiles, K-step 32.
// Weight tile arrives via async global->LDS DMA; x tile is converted in-regs.
// Outputs q,k,v as bf16 in [b, h, n, 32] layout (vectorized b128 stores).
// ---------------------------------------------------------------------------
__global__ __launch_bounds__(256) void qkv_ln_kernel(
    const float* __restrict__ x, const __bf16* __restrict__ W,
    const float* __restrict__ qkv_b,
    const float* __restrict__ kln_w, const float* __restrict__ kln_b,
    const float* __restrict__ vln_w, const float* __restrict__ vln_b,
    __bf16* __restrict__ qbuf, __bf16* __restrict__ kbuf, __bf16* __restrict__ vbuf) {
  __shared__ __align__(16) __bf16 As[128][32];
  __shared__ __align__(16) __bf16 Bs[64][32];
  __shared__ __align__(16) float  Cs[128][66];

  int bx = blockIdx.x;
  int mt = bx / 9, nt = bx % 9;
  int row0 = mt * 128, col0 = nt * 64;
  int tid = threadIdx.x, lane = tid & 31, wave = tid >> 5;
  int wm = wave >> 1, wn = wave & 1;

  v8f acc[2][2] = {};
#pragma unroll 1
  for (int k0 = 0; k0 < CIN; k0 += 32) {
    {  // B tile (weights, already bf16): async DMA straight into LDS
      int r = tid >> 2, qq = (tid & 3) * 8;
      const __bf16* wsrc = W + (size_t)(col0 + r) * CIN + k0 + qq;
      ASYNC_G2L_B128(lds_addr_of(&Bs[r][qq]), wsrc);
    }
    {  // A tile: 128 x 32, f32 -> bf16 conversion in registers
      int r = tid >> 1, hh = (tid & 1) * 16;
      const float* src = x + (size_t)(row0 + r) * CIN + k0 + hh;
      if (k0 + 32 < CIN) __builtin_prefetch(src + 32, 0, 0);
      const float4* s4 = (const float4*)src;
      float4 f0 = s4[0], f1 = s4[1], f2 = s4[2], f3 = s4[3];
      *(v8bf*)&As[r][hh]     = cvt8(f0, f1);
      *(v8bf*)&As[r][hh + 8] = cvt8(f2, f3);
    }
    WAIT_ASYNC0();
    __syncthreads();
    int kb2 = (lane >> 4) * 8, mr = lane & 15;
    v16bf af[2], bfr[2];
#pragma unroll
    for (int i = 0; i < 2; ++i) {
      int m = (wm * 2 + i) * 16 + mr;
      af[i] = pack16(*(const v8bf*)&As[m][kb2], *(const v8bf*)&As[m][16 + kb2]);
    }
#pragma unroll
    for (int j = 0; j < 2; ++j) {
      int n = (wn * 2 + j) * 16 + mr;
      bfr[j] = pack16(*(const v8bf*)&Bs[n][kb2], *(const v8bf*)&Bs[n][16 + kb2]);
    }
#pragma unroll
    for (int i = 0; i < 2; ++i)
#pragma unroll
      for (int j = 0; j < 2; ++j)
        acc[i][j] = WMMA_BF16(af[i], bfr[j], acc[i][j]);
    __syncthreads();
  }

  // Spill accumulators (+bias) to LDS for the LN epilogue.
#pragma unroll
  for (int i = 0; i < 2; ++i)
#pragma unroll
    for (int j = 0; j < 2; ++j)
#pragma unroll
      for (int r = 0; r < 8; ++r) {
        int m = (wm * 2 + i) * 16 + r + ((lane >> 4) << 3);
        int n = (wn * 2 + j) * 16 + (lane & 15);
        Cs[m][n] = acc[i][j][r] + qkv_b[col0 + n];
      }
  __syncthreads();

  // 256 threads -> 128 rows x 2 column-groups of 32 (one head's q, k or v slice).
  int r = tid >> 1, grp = tid & 1;
  int gcol = col0 + grp * 32;
  int head = gcol / 96;
  int kind = (gcol % 96) / 32;  // 0=q, 1=k, 2=v
  size_t token = (size_t)row0 + r;
  int bb = (int)(token >> 16);
  int nn = (int)(token & 65535);

  float vals[32];
#pragma unroll
  for (int d = 0; d < 32; ++d) vals[d] = Cs[r][grp * 32 + d];

  __bf16* dst = qbuf;
  if (kind != 0) {
    float mean = 0.0f;
#pragma unroll
    for (int d = 0; d < 32; ++d) mean += vals[d];
    mean *= (1.0f / 32.0f);
    float var = 0.0f;
#pragma unroll
    for (int d = 0; d < 32; ++d) { float t = vals[d] - mean; var += t * t; }
    var *= (1.0f / 31.0f);  // ddof=1
    float inv = 1.0f / (sqrtf(var) + LN_EPS);
    const float* lw = (kind == 1 ? kln_w : vln_w) + head * 32;
    const float* lb = (kind == 1 ? kln_b : vln_b) + head * 32;
#pragma unroll
    for (int d = 0; d < 32; ++d) vals[d] = (vals[d] - mean) * inv * lw[d] + lb[d];
    dst = (kind == 1) ? kbuf : vbuf;
  }
  size_t o = (((size_t)bb * HEADS + head) * NSEQ + nn) * DH;
  v8bf* dp = (v8bf*)(dst + o);   // 64B-aligned row
#pragma unroll
  for (int g = 0; g < 4; ++g) {
    v8bf pk;
#pragma unroll
    for (int i = 0; i < 8; ++i) pk[i] = (__bf16)vals[g * 8 + i];
    dp[g] = pk;                  // global_store_b128
  }
}

// ---------------------------------------------------------------------------
// Kernel 2: kv[b,h,d,e] = sum_n k[n,d]*v[n,e] / N  via WMMA (A = k transposed
// through LDS), one wave per (pair, N-slice), f32 atomic reduction.
// ---------------------------------------------------------------------------
__global__ __launch_bounds__(32) void kv_kernel(
    const __bf16* __restrict__ kbuf, const __bf16* __restrict__ vbuf,
    float* __restrict__ kv) {
  __shared__ __align__(16) __bf16 kS[32][32];
  __shared__ __align__(16) __bf16 vS[32][32];
  const int SLICES = 64;
  int pair = blockIdx.x / SLICES;
  int slice = blockIdx.x % SLICES;
  int lane = threadIdx.x;
  const __bf16* kb = kbuf + (size_t)pair * NSEQ * DH;
  const __bf16* vb = vbuf + (size_t)pair * NSEQ * DH;
  int tok0 = slice * (NSEQ / SLICES);

  v8f acc[2][2] = {};
#pragma unroll 1
  for (int c = 0; c < NSEQ / SLICES; c += 32) {
    const __bf16* kr = kb + (size_t)(tok0 + c + lane) * DH;
    const __bf16* vr = vb + (size_t)(tok0 + c + lane) * DH;
#pragma unroll
    for (int t = 0; t < 4; ++t) {
      ASYNC_G2L_B128(lds_addr_of(&kS[lane][t * 8]), kr + t * 8);
      ASYNC_G2L_B128(lds_addr_of(&vS[lane][t * 8]), vr + t * 8);
    }
    WAIT_ASYNC0();
    __syncthreads();
    int kb2 = (lane >> 4) * 8, nr = lane & 15;
    v16bf af[2], bfr[2];
#pragma unroll
    for (int i = 0; i < 2; ++i)
#pragma unroll
      for (int j = 0; j < 16; ++j) {
        int kk = ((j >> 3) << 4) + kb2 + (j & 7);   // token index within chunk
        af[i][j]  = kS[kk][i * 16 + nr];            // A[m=d][K=token] (transpose)
        bfr[i][j] = vS[kk][i * 16 + nr];            // B[K=token][n=e]
      }
#pragma unroll
    for (int i = 0; i < 2; ++i)
#pragma unroll
      for (int j = 0; j < 2; ++j)
        acc[i][j] = WMMA_BF16(af[i], bfr[j], acc[i][j]);
    __syncthreads();
  }
  const float scale = 1.0f / (float)NSEQ;
#pragma unroll
  for (int i = 0; i < 2; ++i)
#pragma unroll
    for (int j = 0; j < 2; ++j)
#pragma unroll
      for (int r = 0; r < 8; ++r) {
        int d = i * 16 + r + ((lane >> 4) << 3);
        int e = j * 16 + (lane & 15);
        atomicAdd(&kv[((size_t)pair * 32 + d) * 32 + e], acc[i][j][r] * scale);
      }
}

// ---------------------------------------------------------------------------
// Kernel 3: attn = q @ kv (K=32 -> single WMMA per tile), fused +x residual,
// writes bf16 ret in [token, C] layout.
// ---------------------------------------------------------------------------
__global__ __launch_bounds__(256) void attn_kernel(
    const __bf16* __restrict__ qbuf, const float* __restrict__ kv,
    const float* __restrict__ x, __bf16* __restrict__ ret) {
  int pair = blockIdx.x >> 9;       // / 512 token-tiles
  int tt = blockIdx.x & 511;
  int b = pair / HEADS, h = pair % HEADS;
  int tid = threadIdx.x, lane = tid & 31, wave = tid >> 5;
  int tok0 = tt * 128;
  int kb2 = (lane >> 4) * 8, nr = lane & 15;

  v16bf bfr[2];
#pragma unroll
  for (int j = 0; j < 2; ++j)
#pragma unroll
    for (int jj = 0; jj < 16; ++jj) {
      int kk = ((jj >> 3) << 4) + kb2 + (jj & 7);  // K = d index
      bfr[j][jj] = (__bf16)kv[((size_t)pair * 32 + kk) * 32 + (j * 16 + nr)];
    }

  const __bf16* qrow = qbuf + ((size_t)pair * NSEQ + tok0 + wave * 16 + nr) * DH;
  v16bf af = pack16(*(const v8bf*)(qrow + kb2), *(const v8bf*)(qrow + 16 + kb2));

  v8f acc[2] = {};
  acc[0] = WMMA_BF16(af, bfr[0], acc[0]);
  acc[1] = WMMA_BF16(af, bfr[1], acc[1]);

#pragma unroll
  for (int j = 0; j < 2; ++j)
#pragma unroll
    for (int r = 0; r < 8; ++r) {
      int tokr = tok0 + wave * 16 + r + ((lane >> 4) << 3);
      int e = j * 16 + (lane & 15);
      size_t idx = ((size_t)b * NSEQ + tokr) * CIN + h * DH + e;
      ret[idx] = (__bf16)(acc[j][r] + x[idx]);
    }
}

// ---------------------------------------------------------------------------
// Kernel 4/5: 192x192 GEMM, bf16 A/W, f32 accumulate.
// Tiles arrive via async global->LDS DMA (no VGPR staging).
// EPI=0: exact GELU -> bf16.   EPI=1: +bias +x residual -> f32 output.
// ---------------------------------------------------------------------------
template <int EPI>
__global__ __launch_bounds__(256) void gemm192_kernel(
    const __bf16* __restrict__ A, const __bf16* __restrict__ W,
    const float* __restrict__ bias, const float* __restrict__ xres,
    __bf16* __restrict__ outB, float* __restrict__ outF) {
  __shared__ __align__(16) __bf16 As[128][32];
  __shared__ __align__(16) __bf16 Bs[64][32];

  int bx = blockIdx.x;
  int mt = bx / 3, nt = bx % 3;
  int row0 = mt * 128, col0 = nt * 64;
  int tid = threadIdx.x, lane = tid & 31, wave = tid >> 5;
  int wm = wave >> 1, wn = wave & 1;

  v8f acc[2][2] = {};
#pragma unroll 1
  for (int k0 = 0; k0 < CIN; k0 += 32) {
    {  // A tile 128x32 bf16 = 8KB: two b128 async DMAs per thread
      int r = tid >> 1, hh = (tid & 1) * 16;
      const __bf16* src = A + (size_t)(row0 + r) * CIN + k0 + hh;
      if (k0 + 32 < CIN) __builtin_prefetch(src + 32, 0, 0);
      ASYNC_G2L_B128(lds_addr_of(&As[r][hh]), src);
      ASYNC_G2L_B128(lds_addr_of(&As[r][hh + 8]), src + 8);
    }
    {  // B tile 64x32 bf16 = 4KB: one b128 async DMA per thread
      int r = tid >> 2, qq = (tid & 3) * 8;
      ASYNC_G2L_B128(lds_addr_of(&Bs[r][qq]),
                     W + (size_t)(col0 + r) * CIN + k0 + qq);
    }
    WAIT_ASYNC0();
    __syncthreads();
    int kb2 = (lane >> 4) * 8, mr = lane & 15;
    v16bf af[2], bfr[2];
#pragma unroll
    for (int i = 0; i < 2; ++i) {
      int m = (wm * 2 + i) * 16 + mr;
      af[i] = pack16(*(const v8bf*)&As[m][kb2], *(const v8bf*)&As[m][16 + kb2]);
    }
#pragma unroll
    for (int j = 0; j < 2; ++j) {
      int n = (wn * 2 + j) * 16 + mr;
      bfr[j] = pack16(*(const v8bf*)&Bs[n][kb2], *(const v8bf*)&Bs[n][16 + kb2]);
    }
#pragma unroll
    for (int i = 0; i < 2; ++i)
#pragma unroll
      for (int j = 0; j < 2; ++j)
        acc[i][j] = WMMA_BF16(af[i], bfr[j], acc[i][j]);
    __syncthreads();
  }

#pragma unroll
  for (int i = 0; i < 2; ++i)
#pragma unroll
    for (int j = 0; j < 2; ++j)
#pragma unroll
      for (int r = 0; r < 8; ++r) {
        int m = (wm * 2 + i) * 16 + r + ((lane >> 4) << 3);
        int n = (wn * 2 + j) * 16 + (lane & 15);
        size_t token = (size_t)row0 + m;
        int col = col0 + n;
        float val = acc[i][j][r] + bias[col];
        size_t idx = token * CIN + col;
        if (EPI == 0) {
          float g = 0.5f * val * (1.0f + erff(val * 0.70710678118654752f));
          outB[idx] = (__bf16)g;
        } else {
          outF[idx] = val + xres[idx];
        }
      }
}

// ---------------------------------------------------------------------------
// Launch
// ---------------------------------------------------------------------------
extern "C" void kernel_launch(void* const* d_in, const int* in_sizes, int n_in,
                              void* d_out, int out_size, void* d_ws, size_t ws_size,
                              hipStream_t stream) {
  (void)in_sizes; (void)n_in; (void)out_size; (void)ws_size;
  const float* x     = (const float*)d_in[0];
  const float* qkv_w = (const float*)d_in[1];
  const float* qkv_b = (const float*)d_in[2];
  const float* o1_w  = (const float*)d_in[3];
  const float* o1_b  = (const float*)d_in[4];
  const float* o2_w  = (const float*)d_in[5];
  const float* o2_b  = (const float*)d_in[6];
  const float* kln_w = (const float*)d_in[7];
  const float* kln_b = (const float*)d_in[8];
  const float* vln_w = (const float*)d_in[9];
  const float* vln_b = (const float*)d_in[10];

  char* ws = (char*)d_ws;
  const size_t ACT = (size_t)TOKENS * CIN * sizeof(__bf16);  // 100,663,296 B
  __bf16* qkvw_bf = (__bf16*)(ws + 0);                       //   221,184 B
  __bf16* o1w_bf  = (__bf16*)(ws + 221184);                  //    73,728 B
  __bf16* o2w_bf  = (__bf16*)(ws + 294912);                  //    73,728 B
  float*  kvbuf   = (float*)(ws + 368640);                   //    98,304 B
  __bf16* qbuf    = (__bf16*)(ws + 466944);
  __bf16* kbuf    = (__bf16*)(ws + 466944 + ACT);            // k, later aliased as ret
  __bf16* vbuf    = (__bf16*)(ws + 466944 + 2 * ACT);        // v, later aliased as h

  f32_to_bf16_kernel<<<(CQKV * CIN + 255) / 256, 256, 0, stream>>>(qkv_w, qkvw_bf, CQKV * CIN);
  f32_to_bf16_kernel<<<(CIN * CIN + 255) / 256, 256, 0, stream>>>(o1_w, o1w_bf, CIN * CIN);
  f32_to_bf16_kernel<<<(CIN * CIN + 255) / 256, 256, 0, stream>>>(o2_w, o2w_bf, CIN * CIN);
  zero_f32_kernel<<<(B_ * HEADS * DH * DH + 255) / 256, 256, 0, stream>>>(kvbuf, B_ * HEADS * DH * DH);

  // QKV projection + LN(k,v): 2048 M-tiles x 9 N-tiles
  qkv_ln_kernel<<<(TOKENS / 128) * 9, 256, 0, stream>>>(
      x, qkvw_bf, qkv_b, kln_w, kln_b, vln_w, vln_b, qbuf, kbuf, vbuf);

  // Per-head kv reduction: 24 pairs x 64 slices, one wave each
  kv_kernel<<<B_ * HEADS * 64, 32, 0, stream>>>(kbuf, vbuf, kvbuf);

  // attn = q@kv + x -> ret (aliases kbuf): 24 pairs x 512 token tiles
  attn_kernel<<<B_ * HEADS * 512, 256, 0, stream>>>(qbuf, kvbuf, x, kbuf);

  // h = gelu(ret @ o1^T + b1) -> vbuf
  gemm192_kernel<0><<<(TOKENS / 128) * 3, 256, 0, stream>>>(
      kbuf, o1w_bf, o1_b, nullptr, vbuf, nullptr);

  // out = h @ o2^T + b2 + x -> d_out (f32)
  gemm192_kernel<1><<<(TOKENS / 128) * 3, 256, 0, stream>>>(
      vbuf, o2w_bf, o2_b, x, nullptr, (float*)d_out);
}
